// KV_MoE_plus_75411035783449
// MI455X (gfx1250) — matrix operand order repaired and checked by reading, not compiled, not run on hardware
//
#include <hip/hip_runtime.h>

typedef __attribute__((ext_vector_type(16))) _Float16 v16h;
typedef __attribute__((ext_vector_type(8)))  _Float16 v8h;
typedef __attribute__((ext_vector_type(8)))  float    v8f;

#define CF   256
#define HID  170
#define NE   8
#define NTOK 512
#define K1   2304   // 256 silu + 2048 spline bases
#define K2   1536   // 170 silu + 1360 spline bases + 6 zero pad
#define NP1  176    // 170 outputs padded to 16
#define CAPACITY 161

// ---------------- scalar math helpers ----------------
__device__ __forceinline__ float silu_f(float x){ return x / (1.0f + expf(-x)); }
__device__ __forceinline__ float gelu_f(float x){ return 0.5f * x * (1.0f + erff(x * 0.70710678118654752f)); }

// cubic B-spline bases on the 12-knot grid (knots passed from the reference's grid row)
__device__ __forceinline__ void bspline8(float x, const float* g, float* out){
  float b[11];
#pragma unroll
  for (int j = 0; j < 11; ++j) b[j] = (x >= g[j] && x < g[j+1]) ? 1.0f : 0.0f;
#pragma unroll
  for (int k = 1; k <= 3; ++k){
#pragma unroll
    for (int j = 0; j < 11 - k; ++j){
      float left  = (x - g[j])     / (g[j+k]   - g[j]);
      float right = (g[j+k+1] - x) / (g[j+k+1] - g[j+1]);
      b[j] = left * b[j] + right * b[j+1];   // in-place ascending uses old b[j+1]: correct
    }
  }
#pragma unroll
  for (int j = 0; j < 8; ++j) out[j] = b[j];
}

__device__ __forceinline__ float block_sum(float v, float* smem){ // blockDim.x == 256
  int t = threadIdx.x;
  smem[t] = v; __syncthreads();
#pragma unroll
  for (int s = 128; s > 0; s >>= 1){ if (t < s) smem[t] += smem[t + s]; __syncthreads(); }
  float r = smem[0]; __syncthreads();
  return r;
}

// ---------------- WMMA fragment helpers (CDNA5 16x16x32 f16, wave32) ----------------
// A (16x32, row-major in memory, lda elements): lane<16 -> M=lane, K in {k0..k0+7, k0+16..k0+23}
//                                               lane>=16 -> M=lane-16, K in {k0+8..k0+15, k0+24..k0+31}
__device__ __forceinline__ v16h load_a_frag(const _Float16* __restrict__ A, int lda, int m0, int k0){
  int lane = threadIdx.x & 31;
  const _Float16* p = A + (size_t)(m0 + (lane & 15)) * lda + k0 + ((lane & 16) ? 8 : 0);
  v8h lo = *(const v8h*)p;
  v8h hi = *(const v8h*)(p + 16);
  return __builtin_shufflevector(lo, hi, 0,1,2,3,4,5,6,7,8,9,10,11,12,13,14,15);
}
// B (32x16): B[k][n] = W[n][k] with W stored [out][K] row-major.
// lane holds column n = n0 + (lane&15), K in {k0..k0+7, k0+16..k0+23} (lanes<16) or {k0+8..k0+15, k0+24..k0+31} (lanes>=16)
__device__ __forceinline__ v16h load_b_frag(const _Float16* __restrict__ W, int ldw, int n0, int k0){
  int lane = threadIdx.x & 31;
  const _Float16* p = W + (size_t)(n0 + (lane & 15)) * ldw + k0 + ((lane & 16) ? 8 : 0);
  v8h lo = *(const v8h*)p;
  v8h hi = *(const v8h*)(p + 16);
  return __builtin_shufflevector(lo, hi, 0,1,2,3,4,5,6,7,8,9,10,11,12,13,14,15);
}
__device__ __forceinline__ v8f wmma_f16(v16h a, v16h b, v8f c){
  return __builtin_amdgcn_wmma_f32_16x16x32_f16(false, a, false, b, (short)0, c, false, false);
}

// ---------------- stage 1: 8x8x8 block average pooling ----------------
// wave per (b, gz, gy, c); covers all 4 gx blocks via contiguous float2 row loads
__global__ void kernel_pool(const float* __restrict__ fpn, const float* __restrict__ seg,
                            float* __restrict__ vec){
  int wave = blockIdx.x * 8 + (threadIdx.x >> 5);
  int lane = threadIdx.x & 31;
  int c  = wave & 255;
  int r  = wave >> 8;        // 0..127
  int b  = r >> 4;
  int gz = (r >> 2) & 3;
  int gy = r & 3;
  const float* src = (c < 254) ? (fpn + (size_t)(b * 254 + c) * 32768)
                               : (seg + (size_t)(b * 2 + (c - 254)) * 32768);
  int xoff = 2 * (lane & 15);     // fixed x column pair -> fixed gx
  int yhi  = lane >> 4;
  float acc = 0.0f;
  for (int z8 = 0; z8 < 8; ++z8){
    const float* p = src + (gz * 8 + z8) * 1024 + gy * 256 + xoff;
#pragma unroll
    for (int i = 0; i < 4; ++i){
      float2 v = *(const float2*)(p + (2 * i + yhi) * 32);
      acc += v.x + v.y;
    }
  }
  acc += __shfl_xor(acc, 16);
  acc += __shfl_xor(acc, 1);
  acc += __shfl_xor(acc, 2);
  if ((lane & 3) == 0 && lane < 16){
    int gx = lane >> 2;
    int n = ((b * 4 + gz) * 4 + gy) * 4 + gx;
    vec[(size_t)n * CF + c] = acc * (1.0f / 512.0f);
  }
}

// ---------------- stage 2: layernorm + router + build A1 = [silu(xn) | bs(xn)] (f16) ----------------
__global__ void kernel_router(const float* __restrict__ vec, const float* __restrict__ lnw,
                              const float* __restrict__ lnb, const float* __restrict__ rw,
                              const float* __restrict__ rb, const float* __restrict__ gcf,
                              _Float16* __restrict__ A1, int* __restrict__ top_idx,
                              float* __restrict__ top_w, float* __restrict__ conf){
  __shared__ float smem[256];
  __shared__ float warp_part[64];
  int n = blockIdx.x, t = threadIdx.x;
  float v = vec[(size_t)n * CF + t];
  float mean = block_sum(v, smem) * (1.0f / CF);
  float d = v - mean;
  float var = block_sum(d * d, smem) * (1.0f / CF);
  float xn = d * rsqrtf(var + 1e-5f) * lnw[t] + lnb[t];

  float g[12];
#pragma unroll
  for (int j = 0; j < 12; ++j) g[j] = gcf[j];
  _Float16* row = A1 + (size_t)n * K1;
  row[t] = (_Float16)silu_f(xn);
  float bs[8]; bspline8(xn, g, bs);
#pragma unroll
  for (int j = 0; j < 8; ++j) row[CF + t * 8 + j] = (_Float16)bs[j];

  float p[NE];
#pragma unroll
  for (int e = 0; e < NE; ++e) p[e] = xn * rw[e * CF + t];
#pragma unroll
  for (int e = 0; e < NE; ++e)
    for (int off = 16; off > 0; off >>= 1) p[e] += __shfl_down(p[e], off);
  int lane = t & 31, w = t >> 5;
  if (lane == 0)
    for (int e = 0; e < NE; ++e) warp_part[w * 8 + e] = p[e];
  __syncthreads();
  if (t == 0){
    float sc[NE];
    for (int e = 0; e < NE; ++e){
      float s = rb[e];
      for (int ww = 0; ww < 8; ++ww) s += warp_part[ww * 8 + e];
      sc[e] = s;  // TEMP == 1
    }
    int i0 = 0;
    for (int e = 1; e < NE; ++e) if (sc[e] > sc[i0]) i0 = e;
    int i1 = -1;
    for (int e = 0; e < NE; ++e) if (e != i0 && (i1 < 0 || sc[e] > sc[i1])) i1 = e;
    float v0 = sc[i0], v1 = sc[i1];
    float t1 = expf(v1 - v0);
    float den = 1.0f + t1;
    top_idx[2 * n] = i0; top_idx[2 * n + 1] = i1;
    top_w[2 * n] = 1.0f / den; top_w[2 * n + 1] = t1 / den;
    conf[n] = v0;
  }
}

// ---------------- stage 3: capacity keep mask (wave-scan per expert) ----------------
__global__ void kernel_keep(const int* __restrict__ top_idx, const float* __restrict__ top_w,
                            float* __restrict__ w_keep){
  int e = threadIdx.x >> 5, lane = threadIdx.x & 31;
  int running = 0;
  for (int ch = 0; ch < 16; ++ch){
    int n = ch * 32 + lane;
    int i0 = top_idx[2 * n], i1 = top_idx[2 * n + 1];
    bool sel = (i0 == e) || (i1 == e);
    int incl = sel ? 1 : 0;
#pragma unroll
    for (int off = 1; off < 32; off <<= 1){
      int u = __shfl_up(incl, off);
      if (lane >= off) incl += u;
    }
    int pos = running + incl;
    float w = sel ? ((i0 == e) ? top_w[2 * n] : top_w[2 * n + 1]) : 0.0f;
    w_keep[e * NTOK + n] = (sel && pos <= CAPACITY) ? w : 0.0f;
    running += __shfl(incl, 31);
  }
}

// ---------------- weight prep: fused f16 [base_w | spline_w*scaler] matrices ----------------
__global__ void kernel_w1(const float* __restrict__ bw1, const float* __restrict__ sw1,
                          const float* __restrict__ sc1, _Float16* __restrict__ W1){
  size_t total = (size_t)NE * NP1 * K1;
  for (size_t idx = (size_t)blockIdx.x * blockDim.x + threadIdx.x; idx < total;
       idx += (size_t)gridDim.x * blockDim.x){
    int k = (int)(idx % K1);
    size_t t = idx / K1;
    int o = (int)(t % NP1);
    int e = (int)(t / NP1);
    float v = 0.0f;
    if (o < HID){
      size_t base = (size_t)(e * HID + o) * CF;
      if (k < CF) v = bw1[base + k];
      else { int i = (k - CF) >> 3, j = (k - CF) & 7; v = sw1[(base + i) * 8 + j] * sc1[base + i]; }
    }
    W1[idx] = (_Float16)v;
  }
}
__global__ void kernel_w2(const float* __restrict__ bw2, const float* __restrict__ sw2,
                          const float* __restrict__ sc2, _Float16* __restrict__ W2){
  size_t total = (size_t)NE * CF * K2;
  for (size_t idx = (size_t)blockIdx.x * blockDim.x + threadIdx.x; idx < total;
       idx += (size_t)gridDim.x * blockDim.x){
    int k = (int)(idx % K2);
    size_t t = idx / K2;
    int o = (int)(t % CF);
    int e = (int)(t / CF);
    float v = 0.0f;
    size_t base = (size_t)(e * CF + o) * HID;
    if (k < HID) v = bw2[base + k];
    else if (k < HID + HID * 8){
      int i = (k - HID) >> 3, j = (k - HID) & 7;
      v = sw2[(base + i) * 8 + j] * sc2[base + i];
    }
    W2[idx] = (_Float16)v;
  }
}
__global__ void kernel_padA2(_Float16* __restrict__ A2){
  int idx = blockIdx.x * 256 + threadIdx.x;           // 0..4095 = NE*NTOK rows
  _Float16* row = A2 + (size_t)idx * K2 + (HID + HID * 8); // col 1530
#pragma unroll
  for (int j = 0; j < 6; ++j) row[j] = (_Float16)0.0f;
}

// ---------------- stage 4: expert layer 1 GEMM (WMMA) + gelu + build A2 ----------------
__global__ void kernel_l1(const _Float16* __restrict__ A1, const _Float16* __restrict__ W1,
                          _Float16* __restrict__ A2, const float* __restrict__ ghid){
  int wave = blockIdx.x * 8 + (threadIdx.x >> 5);  // 2816 waves = 8e * 32mt * 11nt
  int e = wave / 352;
  int rem = wave % 352;
  int mt = rem / 11, nt = rem % 11;
  const _Float16* W = W1 + (size_t)e * NP1 * K1;
  v8f c = {};
  for (int k0 = 0; k0 < K1; k0 += 32){
    v16h a = load_a_frag(A1, K1, mt * 16, k0);
    v16h b = load_b_frag(W, K1, nt * 16, k0);
    c = wmma_f16(a, b, c);
  }
  float g[12];
#pragma unroll
  for (int j = 0; j < 12; ++j) g[j] = ghid[j];
  int lane = threadIdx.x & 31;
  int n = nt * 16 + (lane & 15);
  int mbase = mt * 16 + ((lane & 16) ? 8 : 0);
  if (n < HID){
#pragma unroll
    for (int r = 0; r < 8; ++r){
      float h = gelu_f(c[r]);
      _Float16* row = A2 + ((size_t)e * NTOK + (mbase + r)) * K2;
      row[n] = (_Float16)silu_f(h);
      float bs[8]; bspline8(h, g, bs);
#pragma unroll
      for (int j = 0; j < 8; ++j) row[HID + n * 8 + j] = (_Float16)bs[j];
    }
  }
}

// ---------------- stage 5: expert layer 2 GEMM (WMMA), scaled by keep weight ----------------
__global__ void kernel_l2(const _Float16* __restrict__ A2, const _Float16* __restrict__ W2,
                          const float* __restrict__ w_keep, float* __restrict__ ypart){
  int wave = blockIdx.x * 8 + (threadIdx.x >> 5);  // 4096 waves = 8e * 32mt * 16nt
  int e = wave >> 9;
  int rem = wave & 511;
  int mt = rem >> 4, nt = rem & 15;
  const _Float16* A = A2 + (size_t)e * NTOK * K2;
  const _Float16* W = W2 + (size_t)e * CF * K2;
  v8f c = {};
  for (int k0 = 0; k0 < K2; k0 += 32){
    v16h a = load_a_frag(A, K2, mt * 16, k0);
    v16h b = load_b_frag(W, K2, nt * 16, k0);
    c = wmma_f16(a, b, c);
  }
  int lane = threadIdx.x & 31;
  int n = nt * 16 + (lane & 15);
  int mbase = mt * 16 + ((lane & 16) ? 8 : 0);
#pragma unroll
  for (int r = 0; r < 8; ++r){
    int m = mbase + r;
    ypart[((size_t)e * NTOK + m) * CF + n] = c[r] * w_keep[e * NTOK + m];
  }
}

// ---------------- stage 6: deterministic expert sum + layernorm + classifier KAN ----------------
__global__ void kernel_cls(const float* __restrict__ ypart, const float* __restrict__ lnw,
                           const float* __restrict__ lnb, const float* __restrict__ cbw,
                           const float* __restrict__ csw, const float* __restrict__ csc,
                           const float* __restrict__ gcf, float* __restrict__ logits_blk){
  __shared__ float smem[256];
  int n = blockIdx.x, t = threadIdx.x;
  float acc = 0.0f;
#pragma unroll
  for (int e = 0; e < NE; ++e) acc += ypart[((size_t)e * NTOK + n) * CF + t];
  float mean = block_sum(acc, smem) * (1.0f / CF);
  float d = acc - mean;
  float var = block_sum(d * d, smem) * (1.0f / CF);
  float xn = d * rsqrtf(var + 1e-5f) * lnw[t] + lnb[t];
  float g[12];
#pragma unroll
  for (int j = 0; j < 12; ++j) g[j] = gcf[j];
  float bs[8]; bspline8(xn, g, bs);
  float s = silu_f(xn);
  float a0 = 0.0f, a1 = 0.0f;
  {
    float sp = 0.0f;
#pragma unroll
    for (int j = 0; j < 8; ++j) sp += bs[j] * csw[(0 * CF + t) * 8 + j];
    a0 = s * cbw[0 * CF + t] + csc[0 * CF + t] * sp;
  }
  {
    float sp = 0.0f;
#pragma unroll
    for (int j = 0; j < 8; ++j) sp += bs[j] * csw[(1 * CF + t) * 8 + j];
    a1 = s * cbw[1 * CF + t] + csc[1 * CF + t] * sp;
  }
  float r0 = block_sum(a0, smem);
  float r1 = block_sum(a1, smem);
  if (t == 0){ logits_blk[2 * n] = r0; logits_blk[2 * n + 1] = r1; }
}

// ---------------- stage 7: per-batch confidence softmax + weighted sum ----------------
__global__ void kernel_final(const float* __restrict__ conf, const float* __restrict__ logits_blk,
                             float* __restrict__ out){
  __shared__ float smem[64];
  int b = blockIdx.x, t = threadIdx.x;
  float cv = conf[b * 64 + t];
  smem[t] = cv; __syncthreads();
#pragma unroll
  for (int s = 32; s > 0; s >>= 1){ if (t < s) smem[t] = fmaxf(smem[t], smem[t + s]); __syncthreads(); }
  float mx = smem[0]; __syncthreads();
  float ex = expf(cv - mx);
  smem[t] = ex; __syncthreads();
#pragma unroll
  for (int s = 32; s > 0; s >>= 1){ if (t < s) smem[t] += smem[t + s]; __syncthreads(); }
  float S = smem[0]; __syncthreads();
  float wgt = ex / S;
  float l0 = wgt * logits_blk[(b * 64 + t) * 2];
  float l1 = wgt * logits_blk[(b * 64 + t) * 2 + 1];
  smem[t] = l0; __syncthreads();
#pragma unroll
  for (int s = 32; s > 0; s >>= 1){ if (t < s) smem[t] += smem[t + s]; __syncthreads(); }
  float r0 = smem[0]; __syncthreads();
  smem[t] = l1; __syncthreads();
#pragma unroll
  for (int s = 32; s > 0; s >>= 1){ if (t < s) smem[t] += smem[t + s]; __syncthreads(); }
  float r1 = smem[0];
  if (t == 0){ out[2 * b] = r0; out[2 * b + 1] = r1; }
}

// ---------------- launcher ----------------
extern "C" void kernel_launch(void* const* d_in, const int* in_sizes, int n_in,
                              void* d_out, int out_size, void* d_ws, size_t ws_size,
                              hipStream_t stream){
  (void)in_sizes; (void)n_in; (void)out_size; (void)ws_size;
  const float* fpn      = (const float*)d_in[0];
  const float* seg      = (const float*)d_in[1];
  const float* ln_r_w   = (const float*)d_in[2];
  const float* ln_r_b   = (const float*)d_in[3];
  const float* ln_h_w   = (const float*)d_in[4];
  const float* ln_h_b   = (const float*)d_in[5];
  const float* router_w = (const float*)d_in[6];
  const float* router_b = (const float*)d_in[7];
  const float* bw1      = (const float*)d_in[8];
  const float* sw1      = (const float*)d_in[9];
  const float* sc1      = (const float*)d_in[10];
  const float* bw2      = (const float*)d_in[11];
  const float* sw2      = (const float*)d_in[12];
  const float* sc2      = (const float*)d_in[13];
  const float* cls_bw   = (const float*)d_in[14];
  const float* cls_sw   = (const float*)d_in[15];
  const float* cls_sc   = (const float*)d_in[16];
  const float* grid_cf  = (const float*)d_in[17];
  const float* grid_hid = (const float*)d_in[18];
  float* out = (float*)d_out;

  char* ws = (char*)d_ws;
  size_t off = 0;
  auto take = [&](size_t bytes) -> char* {
    off = (off + 255) & ~(size_t)255;
    char* p = ws + off; off += bytes; return p;
  };
  _Float16* A1 = (_Float16*)take((size_t)NTOK * K1 * 2);       // 2.36 MB
  _Float16* W1 = (_Float16*)take((size_t)NE * NP1 * K1 * 2);   // 6.49 MB
  _Float16* A2 = (_Float16*)take((size_t)NE * NTOK * K2 * 2);  // 12.6 MB
  _Float16* W2 = (_Float16*)take((size_t)NE * CF * K2 * 2);    // 6.29 MB
  float* vec        = (float*)take((size_t)NTOK * CF * 4);
  float* ypart      = (float*)take((size_t)NE * NTOK * CF * 4);
  int*   top_idx    = (int*)take(NTOK * 2 * 4);
  float* top_w      = (float*)take(NTOK * 2 * 4);
  float* conf       = (float*)take(NTOK * 4);
  float* w_keep     = (float*)take(NE * NTOK * 4);
  float* logits_blk = (float*)take(NTOK * 2 * 4);

  kernel_pool<<<4096, 256, 0, stream>>>(fpn, seg, vec);
  kernel_router<<<NTOK, 256, 0, stream>>>(vec, ln_r_w, ln_r_b, router_w, router_b,
                                          grid_cf, A1, top_idx, top_w, conf);
  kernel_keep<<<1, 256, 0, stream>>>(top_idx, top_w, w_keep);
  kernel_w1<<<2048, 256, 0, stream>>>(bw1, sw1, sc1, W1);
  kernel_w2<<<2048, 256, 0, stream>>>(bw2, sw2, sc2, W2);
  kernel_padA2<<<16, 256, 0, stream>>>(A2);
  kernel_l1<<<352, 256, 0, stream>>>(A1, W1, A2, grid_hid);
  kernel_l2<<<512, 256, 0, stream>>>(A2, W2, w_keep, ypart);
  kernel_cls<<<NTOK, 256, 0, stream>>>(ypart, ln_h_w, ln_h_b, cls_bw, cls_sw, cls_sc,
                                       grid_cf, logits_blk);
  kernel_final<<<8, 64, 0, stream>>>(conf, logits_blk, out);
}
